// MultiScaleRetention_9698036154508
// MI455X (gfx1250) — matrix operand order repaired and checked by reading, not compiled, not run on hardware
//
#include <hip/hip_runtime.h>

// ---------------- CDNA5 (gfx1250) MultiScaleRetention forward ----------------
// All heavy math via v_wmma_f32_16x16x32_f16 (wave32, f32 accumulate).

typedef __attribute__((ext_vector_type(16))) _Float16 v16h;
typedef __attribute__((ext_vector_type(8)))  _Float16 v8h;
typedef __attribute__((ext_vector_type(8)))  float    v8f;

#define L_SEQ  2048
#define BATCH  2
#define DMODEL 1024
#define NHEAD  8
#define DK     128
#define DV     128
#define ROWS   (BATCH * L_SEQ)   // 4096 total (b,l) rows
#define ODIM   (NHEAD * DV)      // 1024

__device__ __forceinline__ v8f zero8() {
  v8f r;
#pragma unroll
  for (int i = 0; i < 8; ++i) r[i] = 0.0f;
  return r;
}

// Build a v16h fragment from two 16-byte chunks (two ds/global b128 loads).
__device__ __forceinline__ v16h ldfrag(const _Float16* p0, const _Float16* p1) {
  v8h a = *(const v8h*)p0;
  v8h b = *(const v8h*)p1;
  v16h r;
#pragma unroll
  for (int i = 0; i < 8; ++i) { r[i] = a[i]; r[i + 8] = b[i]; }
  return r;
}

__device__ __forceinline__ v8f wmma_f16(v16h a, v16h b, v8f c) {
  // (neg_a, A, neg_b, B, c_mod, C, reuse_a, reuse_b)
  return __builtin_amdgcn_wmma_f32_16x16x32_f16(false, a, false, b, (short)0, c,
                                                false, false);
}

// ---------------------------------------------------------------- converts --
__global__ void f32_to_f16_kernel(const float* __restrict__ in,
                                  _Float16* __restrict__ out, int n) {
  int i = blockIdx.x * blockDim.x + threadIdx.x;
  if (i < n) out[i] = (_Float16)in[i];
}

// in: (H, D, N) f32 row-major  ->  out: (H, N, D) f16 (transposed per head)
__global__ void transposeW_kernel(const float* __restrict__ in,
                                  _Float16* __restrict__ out,
                                  int Hh, int Dd, int Nn) {
  int i = blockIdx.x * blockDim.x + threadIdx.x;
  int total = Hh * Dd * Nn;
  if (i >= total) return;
  int n = i % Nn;
  int d = (i / Nn) % Dd;
  int h = i / (Nn * Dd);
  out[((size_t)h * Nn + n) * Dd + d] = (_Float16)in[i];
}

// -------------------------------------------------------------- WMMA GEMM ---
// C[M x N] = A[M x 1024] * Wt[N x 1024]^T ; A,Wt f16 row-major; K fixed 1024.
// One wave -> 16x64 output tile. Block = 256 threads = 8 waves = 128 rows.
// MODE 0: f16 row-major out (ldc)     MODE 1: f16 transposed out (ld ROWS)
// MODE 2: f32 row-major out (ldc)
template <int MODE>
__global__ __launch_bounds__(256, 1) void gemm_wmma_kernel(
    const _Float16* __restrict__ A, const _Float16* __restrict__ BT,
    void* __restrict__ Cout, int ldc, long bHeadStride, long cHeadStride) {
  const int tid = threadIdx.x;
  const int w   = tid >> 5;
  const int ln  = tid & 31;
  const int nl  = ln & 15;
  const int hi  = ln >> 4;
  const int rowbase = blockIdx.x * 128 + w * 16;
  const int nbase   = blockIdx.y * 64;
  const int z       = blockIdx.z;

  const _Float16* Bz   = BT + (size_t)z * bHeadStride;
  const _Float16* Arow = A + (size_t)(rowbase + nl) * DMODEL;

  v8f acc[4];
#pragma unroll
  for (int n = 0; n < 4; ++n) acc[n] = zero8();

  for (int kc = 0; kc < DMODEL; kc += 32) {
    const _Float16* ap = Arow + kc + hi * 8;
    v16h aF = ldfrag(ap, ap + 16);
#pragma unroll
    for (int n = 0; n < 4; ++n) {
      const _Float16* bp = Bz + (size_t)(nbase + n * 16 + nl) * DMODEL + kc + hi * 8;
      v16h bF = ldfrag(bp, bp + 16);
      acc[n] = wmma_f16(aF, bF, acc[n]);
    }
  }

  if (MODE == 1) {
    _Float16* C = (_Float16*)Cout + (size_t)z * cHeadStride;
#pragma unroll
    for (int n = 0; n < 4; ++n)
#pragma unroll
      for (int r = 0; r < 8; ++r)
        C[(size_t)(nbase + n * 16 + nl) * ROWS + rowbase + r + 8 * hi] =
            (_Float16)acc[n][r];
  } else if (MODE == 0) {
    _Float16* C = (_Float16*)Cout + (size_t)z * cHeadStride;
#pragma unroll
    for (int n = 0; n < 4; ++n)
#pragma unroll
      for (int r = 0; r < 8; ++r)
        C[(size_t)(rowbase + r + 8 * hi) * ldc + nbase + n * 16 + nl] =
            (_Float16)acc[n][r];
  } else {
    float* C = (float*)Cout + (size_t)z * cHeadStride;
#pragma unroll
    for (int n = 0; n < 4; ++n)
#pragma unroll
      for (int r = 0; r < 8; ++r)
        C[(size_t)(rowbase + r + 8 * hi) * ldc + nbase + n * 16 + nl] =
            acc[n][r];
  }
}

// ------------------------------------------------------------------- RoPE ---
// q' = cos*q + sin*rot(q) ; k' = cos*k - sin*rot(k) ; rot(x)[2j]=-x[2j+1].
__global__ void rope_kernel(_Float16* __restrict__ q, _Float16* __restrict__ k) {
  int idx = blockIdx.x * blockDim.x + threadIdx.x;  // H*ROWS*64
  int j   = idx & 63;
  int row = (idx >> 6) & (ROWS - 1);
  int h   = idx >> 18;
  int l   = row & (L_SEQ - 1);
  // theta_j = 10000^(-2j/128) = 2^(-log2(1e4)*2j/128)
  float theta = exp2f(-13.287712379549449f * (float)(2 * j) * (1.0f / 128.0f));
  float angle = (float)l * theta;
  float s, c;
  __sincosf(angle, &s, &c);
  size_t base = (((size_t)h * ROWS) + row) * DK + 2 * j;
  float q0 = (float)q[base], q1 = (float)q[base + 1];
  q[base]     = (_Float16)(c * q0 - s * q1);
  q[base + 1] = (_Float16)(c * q1 + s * q0);
  float k0 = (float)k[base], k1 = (float)k[base + 1];
  k[base]     = (_Float16)(c * k0 + s * k1);
  k[base + 1] = (_Float16)(c * k1 - s * k0);
}

// -------------------------------------------------------------- retention ---
// Two waves (split-K over the causal j-range) per (h, b, 16-row tile):
// 4096 waves total, each accumulating its half of the j-chunks into its own
// fp32 buffer (summed in gn_gate -> deterministic). itile folding balances
// chunk counts across blocks. Per 32-wide j-chunk:
//   S = q_i k_j^T (2x4 WMMA), S *= gamma^(i-j) (mask also covers the causal
//   tail half-tile -> branchless), stage f16 scores in LDS to convert
//   C-frag -> A-frag, then O += P @ v (8 WMMA, depth-2 pipelined vB loads;
//   v stored transposed so B-fragments are two contiguous b128 loads).
__global__ __launch_bounds__(256, 1) void retention_kernel(
    const _Float16* __restrict__ qb, const _Float16* __restrict__ kb,
    const _Float16* __restrict__ vt, float* __restrict__ rbuf0,
    float* __restrict__ rbuf1) {
  __shared__ __align__(16) _Float16 plds[8][16][32];  // per-wave P tile
  const int tid = threadIdx.x;
  const int w   = tid >> 5;
  const int ln  = tid & 31;
  const int nl  = ln & 15;
  const int hi  = ln >> 4;

  // Wave id is uniform: pin it (and everything derived) into SGPRs so the
  // j-loop is a scalar branch and WMMA never sits under an exec mask.
  const int wg    = __builtin_amdgcn_readfirstlane(blockIdx.x * 8 + w);
  const int h     = wg >> 9;           // 512 waves per head
  const int b     = (wg >> 8) & 1;     // 256 waves per (h,b)
  const int t     = wg & 255;
  const int split = t & 1;
  const int traw  = t >> 1;            // 0..127
  // fold so consecutive itiles pair short with long causal ranges
  const int itile = (traw & 1) ? (127 - (traw >> 1)) : (traw >> 1);
  const int bL    = b * L_SEQ;
  const float log2g = __log2f(1.0f - exp2f(-5.0f - (float)h));

  // q A-fragments for this 16x128 row tile: resident for the whole j loop.
  const _Float16* qrow = qb + (((size_t)h * ROWS) + bL + itile * 16 + nl) * DK;
  v16h qA[4];
#pragma unroll
  for (int kk = 0; kk < 4; ++kk) {
    const _Float16* p = qrow + kk * 32 + hi * 8;
    qA[kk] = ldfrag(p, p + 16);
  }

  v8f oacc[8];
#pragma unroll
  for (int n = 0; n < 8; ++n) oacc[n] = zero8();

  const _Float16* kbase = kb + ((size_t)h * ROWS + bL) * DK;
  const _Float16* vbase = vt + ((size_t)h * DV + nl) * ROWS + bL;
  const int nj      = itile * 16 + 16;      // exclusive causal bound on j
  const int nchunks = (nj + 31) >> 5;       // 32-wide chunks
  const int clo     = split ? (nchunks >> 1) : 0;
  const int chi     = split ? nchunks : (nchunks >> 1);

  for (int cb = clo; cb < chi; ++cb) {
    const int jb = cb * 32;
    if (cb + 1 < chi)  // speculative prefetch of next chunk's k rows
      __builtin_prefetch(kbase + (size_t)(jb + 32 + nl) * DK, 0, 1);

#pragma unroll
    for (int jj = 0; jj < 2; ++jj) {
      const int j0 = jb + jj * 16;
      // k rows j0+nl are always < L; the decay mask zeroes any j > i.
      const _Float16* krow = kbase + (size_t)(j0 + nl) * DK;
      v16h kB[4];
#pragma unroll
      for (int kk = 0; kk < 4; ++kk) {
        const _Float16* p = krow + kk * 32 + hi * 8;
        kB[kk] = ldfrag(p, p + 16);
      }
      v8f s = zero8();
#pragma unroll
      for (int kk = 0; kk < 4; ++kk) s = wmma_f16(qA[kk], kB[kk], s);
#pragma unroll
      for (int r = 0; r < 8; ++r) {
        int irow = itile * 16 + r + 8 * hi;
        int jcol = j0 + nl;
        float d = (irow >= jcol) ? exp2f((float)(irow - jcol) * log2g) : 0.0f;
        plds[w][r + 8 * hi][jj * 16 + nl] = (_Float16)(s[r] * d);
      }
    }
    // LDS is per-wave here; DS ops are in-order per wave, but make the
    // store->load dependency explicit with the CDNA5 split counter.
    asm volatile("s_wait_dscnt 0x0" ::: "memory");

    const _Float16* pr = &plds[w][nl][hi * 16];
    v16h pA = ldfrag(pr, pr + 8);

    // v B-fragments, depth-2 pipelined so loads run ahead of their WMMA.
    const _Float16* vp0 = vbase + jb + hi * 16;
    v16h vBa = ldfrag(vp0, vp0 + 8);
    const _Float16* vp1 = vp0 + (size_t)16 * ROWS;
    v16h vBb = ldfrag(vp1, vp1 + 8);
#pragma unroll
    for (int n = 0; n < 8; ++n) {
      v16h cur = vBa;
      vBa = vBb;
      if (n + 2 < 8) {
        const _Float16* vp = vp0 + (size_t)(16 * (n + 2)) * ROWS;
        vBb = ldfrag(vp, vp + 8);
      }
      oacc[n] = wmma_f16(pA, cur, oacc[n]);
    }
    asm volatile("" ::: "memory");  // keep next chunk's LDS stores below
  }

  // partial-sum buffer per split; layout [b*L + l][h][v]
  float* outp = split ? rbuf1 : rbuf0;
#pragma unroll
  for (int n = 0; n < 8; ++n)
#pragma unroll
    for (int r = 0; r < 8; ++r) {
      int i = itile * 16 + r + 8 * hi;
      outp[(((size_t)(bL + i)) * NHEAD + h) * DV + n * 16 + nl] = oacc[n][r];
    }
}

// ------------------------------------------------ groupnorm + SiLU gating ---
__global__ void gn_gate_kernel(const float* __restrict__ rbuf0,
                               const float* __restrict__ rbuf1,
                               const float* __restrict__ gbuf,
                               const float* __restrict__ gnw,
                               const float* __restrict__ gnb,
                               _Float16* __restrict__ gated) {
  const int row  = blockIdx.x;        // b*L + l
  const int h    = threadIdx.x >> 5;  // wave per head
  const int lane = threadIdx.x & 31;
  const size_t base = ((size_t)row * NHEAD + h) * DV;
  float v[4], sum = 0.0f, sq = 0.0f;
#pragma unroll
  for (int i = 0; i < 4; ++i) {
    int c = lane + 32 * i;
    v[i] = rbuf0[base + c] + rbuf1[base + c];  // combine split-K partials
    sum += v[i];
    sq  += v[i] * v[i];
  }
#pragma unroll
  for (int m = 16; m >= 1; m >>= 1) {
    sum += __shfl_xor(sum, m, 32);
    sq  += __shfl_xor(sq, m, 32);
  }
  float mean = sum * (1.0f / 128.0f);
  float var  = sq * (1.0f / 128.0f) - mean * mean;
  float rstd = rsqrtf(var + 1e-5f);
#pragma unroll
  for (int i = 0; i < 4; ++i) {
    int c  = lane + 32 * i;
    int oc = h * DV + c;
    float rn   = (v[i] - mean) * rstd * gnw[oc] + gnb[oc];
    float g    = gbuf[(size_t)row * ODIM + oc];
    float gate = g / (1.0f + __expf(-g));  // g * sigmoid(g)
    gated[(size_t)row * ODIM + oc] = (_Float16)(gate * rn);
  }
}

// ------------------------------------------------------------------ launch --
extern "C" void kernel_launch(void* const* d_in, const int* in_sizes, int n_in,
                              void* d_out, int out_size, void* d_ws,
                              size_t ws_size, hipStream_t stream) {
  const float* seq = (const float*)d_in[0];
  // d_in[1] = times (arange(L), implied by index math) -- unused
  const float* w_q = (const float*)d_in[2];
  const float* w_k = (const float*)d_in[3];
  const float* w_v = (const float*)d_in[4];
  const float* w_g = (const float*)d_in[5];
  const float* w_o = (const float*)d_in[6];
  const float* gnw = (const float*)d_in[7];
  const float* gnb = (const float*)d_in[8];
  float* out = (float*)d_out;

  // workspace bump allocator (256-B aligned)
  char* ws = (char*)d_ws;
  size_t off = 0;
  auto alloc = [&](size_t bytes) -> char* {
    char* p = ws + off;
    off += (bytes + 255) & ~(size_t)255;
    return p;
  };
  _Float16* Xh   = (_Float16*)alloc((size_t)ROWS * DMODEL * 2);      // seq f16
  _Float16* wqT  = (_Float16*)alloc((size_t)NHEAD * DK * DMODEL * 2);
  _Float16* wkT  = (_Float16*)alloc((size_t)NHEAD * DK * DMODEL * 2);
  _Float16* wvT  = (_Float16*)alloc((size_t)NHEAD * DV * DMODEL * 2);
  _Float16* wgT  = (_Float16*)alloc((size_t)ODIM * DMODEL * 2);
  _Float16* woT  = (_Float16*)alloc((size_t)ODIM * ODIM * 2);
  _Float16* qb   = (_Float16*)alloc((size_t)NHEAD * ROWS * DK * 2);
  _Float16* kb   = (_Float16*)alloc((size_t)NHEAD * ROWS * DK * 2);
  _Float16* vt   = (_Float16*)alloc((size_t)NHEAD * DV * ROWS * 2);  // v^T
  float*    gbf  = (float*)alloc((size_t)ROWS * ODIM * 4);
  float*    rbf0 = (float*)alloc((size_t)ROWS * ODIM * 4);
  float*    rbf1 = (float*)alloc((size_t)ROWS * ODIM * 4);
  _Float16* gtd  = (_Float16*)alloc((size_t)ROWS * ODIM * 2);
  (void)ws_size; (void)in_sizes; (void)n_in; (void)out_size;

  // 1) convert inputs to f16 (weights transposed for B-fragment locality)
  {
    int n = ROWS * DMODEL;
    f32_to_f16_kernel<<<n / 256, 256, 0, stream>>>(seq, Xh, n);
  }
  transposeW_kernel<<<(NHEAD * DMODEL * DK) / 256, 256, 0, stream>>>(
      w_q, wqT, NHEAD, DMODEL, DK);
  transposeW_kernel<<<(NHEAD * DMODEL * DK) / 256, 256, 0, stream>>>(
      w_k, wkT, NHEAD, DMODEL, DK);
  transposeW_kernel<<<(NHEAD * DMODEL * DV) / 256, 256, 0, stream>>>(
      w_v, wvT, NHEAD, DMODEL, DV);
  transposeW_kernel<<<(DMODEL * ODIM) / 256, 256, 0, stream>>>(
      w_g, wgT, 1, DMODEL, ODIM);
  transposeW_kernel<<<(ODIM * ODIM) / 256, 256, 0, stream>>>(
      w_o, woT, 1, ODIM, ODIM);

  // 2) projections (per-head GEMMs folded into grid.z)
  dim3 gQKV(ROWS / 128, DK / 64, NHEAD);
  gemm_wmma_kernel<0><<<gQKV, 256, 0, stream>>>(
      Xh, wqT, qb, DK, (long)DK * DMODEL, (long)ROWS * DK);
  gemm_wmma_kernel<0><<<gQKV, 256, 0, stream>>>(
      Xh, wkT, kb, DK, (long)DK * DMODEL, (long)ROWS * DK);
  gemm_wmma_kernel<1><<<gQKV, 256, 0, stream>>>(
      Xh, wvT, vt, 0, (long)DV * DMODEL, (long)DV * ROWS);   // v transposed
  dim3 gG(ROWS / 128, ODIM / 64, 1);
  gemm_wmma_kernel<2><<<gG, 256, 0, stream>>>(Xh, wgT, gbf, ODIM, 0, 0);

  // 3) RoPE on q, k
  rope_kernel<<<(NHEAD * ROWS * 64) / 256, 256, 0, stream>>>(qb, kb);

  // 4) causal decayed retention: 2-way split-K -> 4096 waves, 512 blocks
  retention_kernel<<<(NHEAD * BATCH * (L_SEQ / 16) * 2) / 8, 256, 0, stream>>>(
      qb, kb, vt, rbf0, rbf1);

  // 5) combine split-K partials + groupnorm + SiLU gate -> f16
  gn_gate_kernel<<<ROWS, 256, 0, stream>>>(rbf0, rbf1, gbf, gnw, gnb, gtd);

  // 6) output projection -> fp32 d_out
  gemm_wmma_kernel<2><<<gG, 256, 0, stream>>>(gtd, woT, out, ODIM, 0, 0);
}
